// MultiHeadSelfAttention_88304527605992
// MI455X (gfx1250) — matrix operand order repaired and checked by reading, not compiled, not run on hardware
//
#include <hip/hip_runtime.h>
#include <hip/hip_bf16.h>

// ---------------------------------------------------------------------------
// MHA forward for gfx1250 (MI455X): fp32 in/out, bf16 WMMA internally.
// D_MODEL=1024, H=16, DH=64, B=2, S=2048.
// GEMMs stage weight tiles through LDS with the Tensor Data Mover (TDM),
// double-buffered, with s_wait_tensorcnt synchronization; the staged tile is
// read back with inline-asm ds_load_b128 (explicit LDS addressing).
// ---------------------------------------------------------------------------

typedef __attribute__((ext_vector_type(16))) __bf16 v16bf;
typedef __attribute__((ext_vector_type(8)))  __bf16 v8bf;
typedef __attribute__((ext_vector_type(8)))  float  v8f;
typedef __attribute__((ext_vector_type(4)))  unsigned int u32x4;
typedef __attribute__((ext_vector_type(8)))  int i32x8;
typedef __attribute__((ext_vector_type(4)))  int i32x4;
typedef __attribute__((ext_vector_type(4)))  int v4i;

#define D_MODEL 1024
#define SEQ     2048
#define NHEAD   16
#define DHEAD   64
#define BATCH   2
#define MTOT    (BATCH * SEQ)   // 4096 rows in all big GEMMs
#define KSTEPS  (D_MODEL / 32)  // 32 k-steps of 32

// ---- TDM availability probe (device pass only; host pass uses fallback) ----
#if defined(__AMDGCN__) && __has_builtin(__builtin_amdgcn_tensor_load_to_lds)
  #define HAVE_TDM 1
  #if __has_include(<hip/amd_detail/amd_gfx1250_TDM.h>)
    #define TDM_6ARG 1
  #endif
#endif

__device__ __forceinline__ unsigned short f2bf(float f) {
    unsigned int u = __float_as_uint(f);
    unsigned int r = u + 0x7FFFu + ((u >> 16) & 1u);   // round-to-nearest-even
    return (unsigned short)(r >> 16);
}

__device__ __forceinline__ v16bf cat8(v8bf lo, v8bf hi) {
    return __builtin_shufflevector(lo, hi, 0, 1, 2, 3, 4, 5, 6, 7,
                                           8, 9, 10, 11, 12, 13, 14, 15);
}

// ---------------------------------------------------------------------------
// f32 -> bf16 elementwise (vectorized x4)
// ---------------------------------------------------------------------------
__global__ void __launch_bounds__(256) cvt_kernel(const float* __restrict__ in,
                                                  unsigned short* __restrict__ out,
                                                  int n4) {
    int i = blockIdx.x * blockDim.x + threadIdx.x;
    if (i < n4) {
        float4 f = ((const float4*)in)[i];
        ushort4 o;
        o.x = f2bf(f.x); o.y = f2bf(f.y); o.z = f2bf(f.z); o.w = f2bf(f.w);
        ((ushort4*)out)[i] = o;
    }
}

// ---------------------------------------------------------------------------
// f32 [1024 x 1024] -> transposed bf16 (Wt[n][k] = W[k][n]), LDS tiled
// ---------------------------------------------------------------------------
__global__ void __launch_bounds__(256) tcvt_kernel(const float* __restrict__ in,
                                                   unsigned short* __restrict__ out) {
    __shared__ float tile[32][33];
    const int bx = blockIdx.x & 31;
    const int by = blockIdx.x >> 5;
    const int tx = threadIdx.x & 31;
    const int ty = threadIdx.x >> 5;          // 0..7
#pragma unroll
    for (int r = 0; r < 4; ++r) {
        int row = ty + r * 8;
        tile[row][tx] = in[(size_t)(by * 32 + row) * D_MODEL + bx * 32 + tx];
    }
    __syncthreads();
#pragma unroll
    for (int r = 0; r < 4; ++r) {
        int row = ty + r * 8;
        out[(size_t)(bx * 32 + row) * D_MODEL + by * 32 + tx] = f2bf(tile[tx][row]);
    }
}

// ---------------------------------------------------------------------------
// TDM issue: stage one 32(n) x 32(k) bf16 tile of Bt (row stride 1024 elems)
// into LDS at byte offset lds_off, packed row-major (64 B per row).
// D# built per CDNA5 ISA ch.8: group0 = {count, lds_addr, global_addr, type=2},
// group1 = {data_size=2B, tensor_dims=1024x1024, tile=32x32, stride=1024}.
// ---------------------------------------------------------------------------
#if defined(HAVE_TDM)
__device__ __forceinline__ void tdm_issue_b(unsigned lds_off,
                                            const unsigned short* gsrc) {
    unsigned long long ga = (unsigned long long)(uintptr_t)gsrc;
    u32x4 g0;
    g0[0] = 1u;                                        // count=1, user mode
    g0[1] = lds_off;                                   // LDS byte address
    g0[2] = (unsigned)(ga & 0xFFFFFFFFu);              // global_addr[31:0]
    g0[3] = (unsigned)((ga >> 32) & 0x1FFFFFFu)        // global_addr[56:32]
            | (2u << 30);                              // type=2 ("image")
    i32x8 g1;
    g1[0] = 0x10000;                                   // data_size=1 (2 bytes)
    g1[1] = (int)(1024u << 16);                        // tensor_dim0[15:0]=1024
    g1[2] = (int)(1024u << 16);                        // tensor_dim1[15:0]=1024
    g1[3] = (int)(32u << 16);                          // tile_dim0=32
    g1[4] = 32;                                        // tile_dim1=32
    g1[5] = 1024;                                      // tensor_dim0_stride=1024
    g1[6] = 0;
    g1[7] = 0;
    i32x4 gz = {0, 0, 0, 0};
#if defined(TDM_6ARG)
    i32x8 gz8 = {0, 0, 0, 0, 0, 0, 0, 0};
    __builtin_amdgcn_tensor_load_to_lds(g0, g1, gz, gz, gz8, 0);
#else
    __builtin_amdgcn_tensor_load_to_lds(g0, g1, gz, gz, 0);
#endif
}
#else
// Fallback: 256 threads cooperatively copy the 2 KB tile (8 B per thread).
__device__ __forceinline__ void stage_b_fallback(unsigned short* dst,
                                                 const unsigned short* src) {
    const int t = threadIdx.x;
    const int row = t >> 3, ch = t & 7;
    *(unsigned long long*)(dst + row * 32 + ch * 4) =
        *(const unsigned long long*)(src + (size_t)row * D_MODEL + ch * 4);
}
#endif

// ---------------------------------------------------------------------------
// GEMM: C[4096,1024] = A(bf16,[4096,1024]) * Bt^T + bias
//   Bt stored transposed: Bt[n][k]  (contiguous along k)
//   mode 0: store bf16 row-major          (Q, K)
//   mode 1: store bf16 as Vt[b][h][dh][s] (head-transposed V)
//   mode 2: store f32 row-major           (final output)
// Block = 256 threads = 8 waves; block tile 256(m) x 32(n); all 8 waves share
// one TDM-staged, double-buffered B tile per 32-wide k-step. Each wave owns a
// 32x32 C tile (2x2 v_wmma_f32_16x16x32_bf16). A fragments software-pipelined.
//
// NOTE on the B-fragment reads: the TDM writes LDS through a D# descriptor
// the compiler cannot see, so plain loads get folded to undef (round 2) and
// volatile loads lose AS3 provenance and become serialized SYS-scope flat
// loads (round 3). Inline-asm ds_load_b128 with explicitly computed LDS byte
// addresses (btile is the sole __shared__ object -> base offset 0) emits the
// intended DS path and is opaque to the optimizer.
// ---------------------------------------------------------------------------
__global__ void __launch_bounds__(256) gemm_kernel(const unsigned short* __restrict__ A,
                                                   const unsigned short* __restrict__ Bt,
                                                   const float* __restrict__ bias,
                                                   void* __restrict__ Out,
                                                   int mode) {
    __shared__ unsigned short btile[2][32 * 32];   // 2 x 2 KB, LDS offset 0 / 2048
    const int lane = threadIdx.x & 31;
    const int wib  = threadIdx.x >> 5;             // 0..7
    const int l15  = lane & 15;
    const int hi   = lane >> 4;

    const int ntiles = D_MODEL / 32;               // 32
    const int mb = (blockIdx.x / ntiles) * 256 + wib * 32;
    const int nb = (blockIdx.x % ntiles) * 32;

    v8f acc[2][2];
#pragma unroll
    for (int i = 0; i < 2; ++i)
#pragma unroll
        for (int j = 0; j < 2; ++j) acc[i][j] = {};

    float bias_n[2];
#pragma unroll
    for (int jn = 0; jn < 2; ++jn) bias_n[jn] = bias[nb + jn * 16 + l15];

    // ---- stage first B tile ----
#if defined(HAVE_TDM)
    if (threadIdx.x < 32) tdm_issue_b(0u, Bt + (size_t)nb * D_MODEL);
#else
    stage_b_fallback(&btile[0][0], Bt + (size_t)nb * D_MODEL);
#endif

    // ---- preload A fragments for kc = 0 ----
    v16bf a[2];
#pragma unroll
    for (int im = 0; im < 2; ++im) {
        const unsigned short* ar = A + (size_t)(mb + im * 16 + l15) * D_MODEL + hi * 16;
        a[im] = cat8(*(const v8bf*)(ar), *(const v8bf*)(ar + 8));
    }

    for (int s = 0; s < KSTEPS; ++s) {
        const int kc  = s * 32;
        const int cur = s & 1;

        // issue next tile's DMA, then wait until the current tile has landed
#if defined(HAVE_TDM)
        if (threadIdx.x < 32) {
            if (s + 1 < KSTEPS) {
                tdm_issue_b((unsigned)((cur ^ 1) * 2048),
                            Bt + (size_t)nb * D_MODEL + (kc + 32));
                __builtin_amdgcn_s_wait_tensorcnt(1);
            } else {
                __builtin_amdgcn_s_wait_tensorcnt(0);
            }
        }
#else
        if (s + 1 < KSTEPS)
            stage_b_fallback(&btile[cur ^ 1][0],
                             Bt + (size_t)nb * D_MODEL + (kc + 32));
#endif
        __syncthreads();   // publish btile[cur] to all 8 waves

        // B fragments from LDS (32x16 each, 16 contiguous K per half-wave)
        v16bf bf[2];
#if defined(HAVE_TDM)
        {
            // byte addr inside wave's LDS allocation (btile at offset 0)
            const unsigned a0 = (unsigned)(cur * 2048 + l15 * 64 + hi * 32);
            const unsigned a1 = a0 + 1024;                 // second n-frag (+16 rows)
            v4i b0l, b0h, b1l, b1h;
            asm volatile("ds_load_b128 %0, %4\n\t"
                         "ds_load_b128 %1, %4 offset:16\n\t"
                         "ds_load_b128 %2, %5\n\t"
                         "ds_load_b128 %3, %5 offset:16\n\t"
                         "s_wait_dscnt 0"
                         : "=&v"(b0l), "=&v"(b0h), "=&v"(b1l), "=&v"(b1h)
                         : "v"(a0), "v"(a1));
            bf[0] = cat8(__builtin_bit_cast(v8bf, b0l), __builtin_bit_cast(v8bf, b0h));
            bf[1] = cat8(__builtin_bit_cast(v8bf, b1l), __builtin_bit_cast(v8bf, b1h));
        }
#else
#pragma unroll
        for (int in_ = 0; in_ < 2; ++in_)
            bf[in_] = *(const v16bf*)(&btile[cur][(in_ * 16 + l15) * 32 + hi * 16]);
#endif

        // software-pipeline next A fragments
        v16bf an[2];
        if (s + 1 < KSTEPS) {
#pragma unroll
            for (int im = 0; im < 2; ++im) {
                const unsigned short* ar =
                    A + (size_t)(mb + im * 16 + l15) * D_MODEL + (kc + 32) + hi * 16;
                __builtin_prefetch(ar + 32, 0, 1);
                an[im] = cat8(*(const v8bf*)(ar), *(const v8bf*)(ar + 8));
            }
        }

#pragma unroll
        for (int im = 0; im < 2; ++im)
#pragma unroll
            for (int in_ = 0; in_ < 2; ++in_)
                acc[im][in_] = __builtin_amdgcn_wmma_f32_16x16x32_bf16(
                    false, a[im], false, bf[in_], (short)0, acc[im][in_], false, false);

        a[0] = an[0];
        a[1] = an[1];
        __syncthreads();   // all waves done with btile[cur] before it is re-DMAed
    }

#pragma unroll
    for (int im = 0; im < 2; ++im)
#pragma unroll
        for (int in_ = 0; in_ < 2; ++in_)
#pragma unroll
            for (int v = 0; v < 8; ++v) {
                const int m = mb + im * 16 + v + hi * 8;
                const int n = nb + in_ * 16 + l15;
                const float val = acc[im][in_][v] + bias_n[in_];
                if (mode == 0) {
                    ((unsigned short*)Out)[(size_t)m * D_MODEL + n] = f2bf(val);
                } else if (mode == 1) {
                    const int hh = n >> 6, dh = n & 63;
                    const int bb = m >> 11, sl = m & (SEQ - 1);
                    ((unsigned short*)Out)[(size_t)((bb * NHEAD + hh) * DHEAD + dh) * SEQ + sl]
                        = f2bf(val);
                } else {
                    ((float*)Out)[(size_t)m * D_MODEL + n] = val;
                }
            }
}

// ---------------------------------------------------------------------------
// Causal flash attention. One wave = one 16-row tile of one (b, h).
//   Qb, Kb: bf16 [B*S, 1024] row-major;  Vt: bf16 [B, H, DH, S]
//   vals:   bf16 [B*S, 1024] (input of the output projection)
// j-loop step 32: 4 WMMAs for scores, online softmax via __shfl_xor row
// reductions, P re-layout through per-wave LDS (in-order LDS + s_wait_dscnt),
// then 4 WMMAs for P(16x32) x V(32x64).
// ---------------------------------------------------------------------------
__global__ void __launch_bounds__(256) attn_kernel(const unsigned short* __restrict__ Qb,
                                                   const unsigned short* __restrict__ Kb,
                                                   const unsigned short* __restrict__ Vt,
                                                   unsigned short* __restrict__ vals) {
    __shared__ unsigned short plds[8 * 16 * 32];   // 1 KB per wave
    const int lane = threadIdx.x & 31;
    const int wib  = threadIdx.x >> 5;
    const int wid  = blockIdx.x * 8 + wib;
    const int l15  = lane & 15;
    const int hi   = lane >> 4;

    const int mtiles = SEQ / 16;                   // 128
    const int b  = wid / (NHEAD * mtiles);
    const int hm = wid % (NHEAD * mtiles);
    const int h  = hm / mtiles;
    const int mb = (hm % mtiles) * 16;

    unsigned short* pw = &plds[wib * 512];

    // Q A-fragments for Kc ranges [0,32) and [32,64)
    const unsigned short* qrow =
        Qb + (size_t)(b * SEQ + mb + l15) * D_MODEL + h * DHEAD + hi * 16;
    v16bf qa[2];
#pragma unroll
    for (int c = 0; c < 2; ++c) {
        v8bf lo  = *(const v8bf*)(qrow + c * 32);
        v8bf hi8 = *(const v8bf*)(qrow + c * 32 + 8);
        qa[c] = cat8(lo, hi8);
    }

    v8f acc[4];
#pragma unroll
    for (int t = 0; t < 4; ++t) acc[t] = {};
    float rmax[8], rsum[8];
#pragma unroll
    for (int v = 0; v < 8; ++v) { rmax[v] = -3.0e38f; rsum[v] = 0.0f; }

    const float scale = 0.125f;                    // 1/sqrt(64)
    const int jend = mb + 16;                      // causal bound

    for (int jb = 0; jb < jend; jb += 32) {
        // ---- scores: two 16-col groups ----
        v8f sc[2];
#pragma unroll
        for (int jt = 0; jt < 2; ++jt) {
            const unsigned short* krow =
                Kb + (size_t)(b * SEQ + jb + jt * 16 + l15) * D_MODEL + h * DHEAD + hi * 16;
            v16bf kb0 = *(const v16bf*)(krow);
            v16bf kb1 = *(const v16bf*)(krow + 32);
            v8f c = {};
            c = __builtin_amdgcn_wmma_f32_16x16x32_bf16(false, qa[0], false, kb0,
                                                        (short)0, c, false, false);
            c = __builtin_amdgcn_wmma_f32_16x16x32_bf16(false, qa[1], false, kb1,
                                                        (short)0, c, false, false);
            sc[jt] = c;
        }

        // ---- online softmax over the 32 new columns; write P to LDS ----
#pragma unroll
        for (int v = 0; v < 8; ++v) {
            const int gi = mb + v + hi * 8;        // global row index
            float s0 = sc[0][v] * scale;
            float s1 = sc[1][v] * scale;
            if (jb + l15 > gi)      s0 = -3.0e38f; // causal mask
            if (jb + 16 + l15 > gi) s1 = -3.0e38f;
            float mx = fmaxf(s0, s1);
#pragma unroll
            for (int off = 1; off < 16; off <<= 1)
                mx = fmaxf(mx, __shfl_xor(mx, off));
            const float mnew = fmaxf(rmax[v], mx);
            const float corr = __expf(rmax[v] - mnew);
            rmax[v] = mnew;
            const float p0 = __expf(s0 - mnew);
            const float p1 = __expf(s1 - mnew);
            float ps = p0 + p1;
#pragma unroll
            for (int off = 1; off < 16; off <<= 1)
                ps += __shfl_xor(ps, off);
            rsum[v] = rsum[v] * corr + ps;
#pragma unroll
            for (int t = 0; t < 4; ++t) acc[t][v] *= corr;
            const int m = v + hi * 8;
            pw[m * 32 + l15]      = f2bf(p0);
            pw[m * 32 + 16 + l15] = f2bf(p1);
        }
        asm volatile("s_wait_dscnt 0" ::: "memory");  // LDS stores -> loads

        // ---- reload P as a 16x32 A-fragment ----
        v8bf plo = *(const v8bf*)(&pw[l15 * 32 + hi * 16]);
        v8bf phi = *(const v8bf*)(&pw[l15 * 32 + 8 + hi * 16]);
        v16bf pa = cat8(plo, phi);
        asm volatile("s_wait_dscnt 0" ::: "memory");  // loads done before next stores

        // ---- P x V: 4 dh-tiles of 16 ----
#pragma unroll
        for (int t = 0; t < 4; ++t) {
            const unsigned short* vrow =
                Vt + (size_t)((b * NHEAD + h) * DHEAD + t * 16 + l15) * SEQ + jb + hi * 16;
            v16bf vb = *(const v16bf*)(vrow);
            acc[t] = __builtin_amdgcn_wmma_f32_16x16x32_bf16(false, pa, false, vb,
                                                             (short)0, acc[t], false, false);
        }
    }

    // ---- normalize and store (bf16, row-major merged heads) ----
#pragma unroll
    for (int t = 0; t < 4; ++t)
#pragma unroll
        for (int v = 0; v < 8; ++v) {
            const int m = mb + v + hi * 8;
            const int n = h * DHEAD + t * 16 + l15;
            vals[(size_t)(b * SEQ + m) * D_MODEL + n] = f2bf(acc[t][v] / rsum[v]);
        }
}

// ---------------------------------------------------------------------------
// Launch
// ---------------------------------------------------------------------------
extern "C" void kernel_launch(void* const* d_in, const int* in_sizes, int n_in,
                              void* d_out, int out_size, void* d_ws, size_t ws_size,
                              hipStream_t stream) {
    const float* x  = (const float*)d_in[0];
    const float* Wq = (const float*)d_in[1];
    const float* bq = (const float*)d_in[2];
    const float* Wk = (const float*)d_in[3];
    const float* bk = (const float*)d_in[4];
    const float* Wv = (const float*)d_in[5];
    const float* bv = (const float*)d_in[6];
    const float* Wo = (const float*)d_in[7];
    const float* bo = (const float*)d_in[8];

    const size_t MD = (size_t)MTOT * D_MODEL;      // 4M elements
    const size_t WD = (size_t)D_MODEL * D_MODEL;   // 1M elements

    unsigned short* ws   = (unsigned short*)d_ws;
    unsigned short* xb   = ws;                     // bf16 x            [4096,1024]
    unsigned short* Wqt  = xb  + MD;               // bf16 Wq^T         [1024,1024]
    unsigned short* Wkt  = Wqt + WD;
    unsigned short* Wvt  = Wkt + WD;
    unsigned short* Wot  = Wvt + WD;
    unsigned short* Qb   = Wot + WD;               // bf16 Q            [4096,1024]
    unsigned short* Kb2  = Qb  + MD;               // bf16 K            [4096,1024]
    unsigned short* Vt   = Kb2 + MD;               // bf16 V^T          [B,H,DH,S]
    unsigned short* vals = Vt  + MD;               // bf16 attn output  [4096,1024]

    // 1) precision conversion (+ weight transpose)
    cvt_kernel<<<(int)(MD / 4 / 256), 256, 0, stream>>>(x, xb, (int)(MD / 4));
    tcvt_kernel<<<1024, 256, 0, stream>>>(Wq, Wqt);
    tcvt_kernel<<<1024, 256, 0, stream>>>(Wk, Wkt);
    tcvt_kernel<<<1024, 256, 0, stream>>>(Wv, Wvt);
    tcvt_kernel<<<1024, 256, 0, stream>>>(Wo, Wot);

    // 2) QKV projections (WMMA + TDM-staged weights)
    gemm_kernel<<<512, 256, 0, stream>>>(xb, Wqt, bq, (void*)Qb, 0);
    gemm_kernel<<<512, 256, 0, stream>>>(xb, Wkt, bk, (void*)Kb2, 0);
    gemm_kernel<<<512, 256, 0, stream>>>(xb, Wvt, bv, (void*)Vt, 1);

    // 3) causal flash attention (WMMA)
    attn_kernel<<<512, 256, 0, stream>>>(Qb, Kb2, Vt, vals);

    // 4) output projection -> fp32 d_out (WMMA)
    gemm_kernel<<<512, 256, 0, stream>>>(vals, Wot, bo, d_out, 2);
}
